// WaveGlow_52810917872215
// MI455X (gfx1250) — compile-verified, matrix operand order checked by reading, not run on hardware
//
#include <hip/hip_runtime.h>
#include <hip/hip_bf16.h>
#include <math.h>

// ---------------------------------------------------------------------------
// WaveGlow forward for gfx1250 (MI455X).  Heavy GEMMs run on
// v_wmma_f32_16x16x32_f16.  Each wave owns a 32x64 output tile (2x4 WMMA
// tiles, 8 f32 accumulators): per K-step 12 b128 loads feed 8 WMMAs
// (vs 4:1 for a single-tile wave), ~21 FLOP/byte from L2.
// Activations are time-major [t][ch] f16 so A (weights, row-major) and B^T
// fragments both load as contiguous b128 pairs -- no transpose needed.
// ---------------------------------------------------------------------------

typedef __attribute__((ext_vector_type(16))) _Float16 v16h;
typedef __attribute__((ext_vector_type(8)))  _Float16 half8;
typedef __attribute__((ext_vector_type(8)))  float    v8f;

union H16 { half8 h[2]; v16h v; };

#define NB 2
#define TG 1024
#define CH 256

#define WMMA_F16(a, b, c) \
  __builtin_amdgcn_wmma_f32_16x16x32_f16(false, (a).v, false, (b).v, (short)0, (c), false, false)

// ---- 32x64 wave tile: C[2][4] 16x16 accumulators over K --------------------
// A: MxK row-major f16;  BT: NxK row-major f16 (i.e. activations [t][ch])
__device__ __forceinline__ void wmma_tile_2x4(const _Float16* __restrict__ A, int ldA,
                                              const _Float16* __restrict__ BT, int ldBT,
                                              int M0, int N0, int K, v8f c[2][4]) {
  const int lane = threadIdx.x & 31;
  const int r    = lane & 15;
  const int kh   = (lane >> 4) << 3;   // upper half-wave: K+8..15 / K+24..31
  const _Float16* pa0 = A  + (size_t)(M0 + r) * ldA  + kh;
  const _Float16* pa1 = pa0 + (size_t)16 * ldA;
  const _Float16* pb0 = BT + (size_t)(N0 + r) * ldBT + kh;
  const _Float16* pb1 = pb0 + (size_t)16 * ldBT;
  const _Float16* pb2 = pb0 + (size_t)32 * ldBT;
  const _Float16* pb3 = pb0 + (size_t)48 * ldBT;
  for (int k0 = 0; k0 < K; k0 += 32) {
    H16 a0, a1, b0, b1, b2, b3;
    a0.h[0] = *(const half8*)(pa0 + k0); a0.h[1] = *(const half8*)(pa0 + k0 + 16);
    a1.h[0] = *(const half8*)(pa1 + k0); a1.h[1] = *(const half8*)(pa1 + k0 + 16);
    b0.h[0] = *(const half8*)(pb0 + k0); b0.h[1] = *(const half8*)(pb0 + k0 + 16);
    b1.h[0] = *(const half8*)(pb1 + k0); b1.h[1] = *(const half8*)(pb1 + k0 + 16);
    b2.h[0] = *(const half8*)(pb2 + k0); b2.h[1] = *(const half8*)(pb2 + k0 + 16);
    b3.h[0] = *(const half8*)(pb3 + k0); b3.h[1] = *(const half8*)(pb3 + k0 + 16);
    c[0][0] = WMMA_F16(a0, b0, c[0][0]);
    c[0][1] = WMMA_F16(a0, b1, c[0][1]);
    c[0][2] = WMMA_F16(a0, b2, c[0][2]);
    c[0][3] = WMMA_F16(a0, b3, c[0][3]);
    c[1][0] = WMMA_F16(a1, b0, c[1][0]);
    c[1][1] = WMMA_F16(a1, b1, c[1][1]);
    c[1][2] = WMMA_F16(a1, b2, c[1][2]);
    c[1][3] = WMMA_F16(a1, b3, c[1][3]);
  }
}

// block = 256 threads = 8 waves arranged 4(M) x 2(N): block tile 128M x 128N
__device__ __forceinline__ void wave_tile_origin(int& M0, int& N0) {
  int w = threadIdx.x >> 5;
  M0 = blockIdx.x * 128 + (w >> 1) * 32;
  N0 = blockIdx.y * 128 + (w & 1) * 64;
}

// ---- upsample (ConvTranspose1d 80->80, k=1024, s=256) fused with the
//      (B,80,T)->(B,640,Tg) regroup; writes f16 conditioning s16[b][tg][640]
__global__ void k_upsample(const float* __restrict__ spect, const float* __restrict__ upw,
                           const float* __restrict__ upb, _Float16* __restrict__ s16) {
  int idx = blockIdx.x * blockDim.x + threadIdx.x;
  if (idx >= NB * 80 * 8192) return;
  int t = idx & 8191;
  int m = (idx >> 13) % 80;
  int b = idx / (80 * 8192);
  float acc = upb[m];
  int fmax = t >> 8; if (fmax > 31) fmax = 31;
  int fmin = (t - 768) >> 8; if (fmin < 0) fmin = 0;   // ceil((t-1023)/256)
  for (int f = fmin; f <= fmax; ++f) {
    int kk = t - (f << 8);                              // 0..1023
    float s = 0.f;
    for (int i = 0; i < 80; ++i)
      s += spect[(b * 80 + i) * 32 + f] * upw[((size_t)i * 80 + m) * 1024 + kk];
    acc += s;
  }
  int tg = t >> 3, g = t & 7;
  s16[((size_t)b * TG + tg) * 640 + m * 8 + g] = (_Float16)acc;
}

// ---- helpers -------------------------------------------------------------
__global__ void k_f2h(const float* __restrict__ src, _Float16* __restrict__ dst, int n) {
  int i = blockIdx.x * blockDim.x + threadIdx.x;
  if (i < n) dst[i] = (_Float16)src[i];
}

// in_w [512][256][3] -> f16 [512][768] with K index = j*256 + c
__global__ void k_conv_inw(const float* __restrict__ w, _Float16* __restrict__ w16) {
  int idx = blockIdx.x * blockDim.x + threadIdx.x;
  if (idx >= 512 * 768) return;
  int o = idx / 768, k = idx % 768;
  int j = k >> 8, c = k & 255;
  w16[idx] = (_Float16)w[((size_t)o * 256 + c) * 3 + j];
}

// audio (B,8192) -> a[b][g][tg]
__global__ void k_audio_init(const float* __restrict__ audio, float* __restrict__ a) {
  int idx = blockIdx.x * blockDim.x + threadIdx.x;
  if (idx >= NB * 8 * TG) return;
  int tg = idx & (TG - 1);
  int g  = (idx >> 10) & 7;
  int b  = idx >> 13;
  a[idx] = audio[(size_t)b * 8192 + tg * 8 + g];
}

// invertible 1x1: aOut[o, t] = sum_i W[o][i] * aIn[i, t]   (n <= 8)
__global__ void k_mmw(const float* __restrict__ W, const float* __restrict__ aIn,
                      float* __restrict__ aOut, int n, int chBase) {
  int idx = blockIdx.x * blockDim.x + threadIdx.x;
  if (idx >= NB * n * TG) return;
  int t = idx % TG;
  int o = (idx / TG) % n;
  int b = idx / (TG * n);
  float s = 0.f;
  for (int i = 0; i < n; ++i)
    s += W[o * n + i] * aIn[(size_t)b * 8192 + (chBase + i) * TG + t];
  aOut[(size_t)b * 8192 + (chBase + o) * TG + t] = s;
}

// start 1x1 (K = nh <= 4) -> aCur16[b][t][256]; also zeroes skip accumulator
__global__ void k_start(const float* __restrict__ sw, const float* __restrict__ sb,
                        const float* __restrict__ aIn, int nh, int chBase,
                        _Float16* __restrict__ aCur, float* __restrict__ outAcc) {
  int idx = blockIdx.x * blockDim.x + threadIdx.x;
  if (idx >= NB * TG * CH) return;
  int c = idx & 255;
  int t = (idx >> 8) & (TG - 1);
  int b = idx >> 18;
  float s = sb[c];
  for (int g = 0; g < nh; ++g)
    s += sw[c * nh + g] * aIn[(size_t)b * 8192 + (chBase + g) * TG + t];
  aCur[idx]   = (_Float16)s;
  outAcc[idx] = 0.f;
}

// dilated-conv im2col: imcol[b][t][j*256+c] = aCur[b][t+(j-1)*dil][c] (zero pad)
__global__ void k_im2col(const _Float16* __restrict__ aCur, _Float16* __restrict__ imcol,
                         int dil) {
  int idx = blockIdx.x * blockDim.x + threadIdx.x;
  if (idx >= NB * TG * 768) return;
  int k = idx % 768;
  int t = (idx / 768) % TG;
  int b = idx / (768 * TG);
  int j = k >> 8, c = k & 255;
  int ts = t + (j - 1) * dil;
  _Float16 v = (_Float16)0.f;
  if (ts >= 0 && ts < TG) v = aCur[((size_t)b * TG + ts) * CH + c];
  imcol[idx] = v;
}

// ---- WMMA GEMM kernels ----------------------------------------------------
// cond = cond_w(4096x640) @ s -> f32 cond[b][t][4096]   grid (32,8,NB)x256
__global__ void k_gemm_cond(const _Float16* __restrict__ cw16, const _Float16* __restrict__ s16,
                            const float* __restrict__ cb, float* __restrict__ cond) {
  int M0, N0; wave_tile_origin(M0, N0);
  int b = blockIdx.z;
  v8f c[2][4] = {};
  wmma_tile_2x4(cw16, 640, s16 + (size_t)b * TG * 640, 640, M0, N0, 640, c);
  int lane = threadIdx.x & 31;
  int nl = lane & 15, mh = (lane >> 4) << 3;
#pragma unroll
  for (int ni = 0; ni < 4; ++ni) {
    float* base = cond + ((size_t)b * TG + (N0 + ni * 16 + nl)) * 4096;
#pragma unroll
    for (int mi = 0; mi < 2; ++mi)
#pragma unroll
      for (int r = 0; r < 8; ++r) {
        int m = M0 + mi * 16 + mh + r;
        base[m] = c[mi][ni][r] + cb[m];
      }
  }
}

// in_act = in_w(512x768) @ im2col + in_b + cond_slice -> f32 [b][t][512]
__global__ void k_gemm_inact(const _Float16* __restrict__ w16, const _Float16* __restrict__ imcol,
                             const float* __restrict__ ib, const float* __restrict__ cond,
                             int condOff, float* __restrict__ inAct) {
  int M0, N0; wave_tile_origin(M0, N0);
  int b = blockIdx.z;
  v8f c[2][4] = {};
  wmma_tile_2x4(w16, 768, imcol + (size_t)b * TG * 768, 768, M0, N0, 768, c);
  int lane = threadIdx.x & 31;
  int nl = lane & 15, mh = (lane >> 4) << 3;
#pragma unroll
  for (int ni = 0; ni < 4; ++ni) {
    int t = N0 + ni * 16 + nl;
    const float* cbase = cond + ((size_t)b * TG + t) * 4096 + condOff;
    float* obase = inAct + ((size_t)b * TG + t) * 512;
#pragma unroll
    for (int mi = 0; mi < 2; ++mi)
#pragma unroll
      for (int r = 0; r < 8; ++r) {
        int m = M0 + mi * 16 + mh + r;
        obase[m] = c[mi][ni][r] + ib[m] + cbase[m];
      }
  }
}

// gated activation: acts = tanh(in_act[:256]) * sigmoid(in_act[256:])  (f16 out)
__global__ void k_act(const float* __restrict__ inAct, _Float16* __restrict__ acts) {
  int idx = blockIdx.x * blockDim.x + threadIdx.x;
  if (idx >= NB * TG * CH) return;
  int c = idx & 255;
  int t = (idx >> 8) & (TG - 1);
  int b = idx >> 18;
  const float* row = inAct + ((size_t)b * TG + t) * 512;
  float ta = tanhf(row[c]);
  float sg = 1.f / (1.f + expf(-row[c + 256]));
  acts[idx] = (_Float16)(ta * sg);
}

// res/skip GEMM with fused residual update:
//   rows <256 -> aCur += ; rows >=256 (or all on last layer) -> outAcc +=
__global__ void k_gemm_rs(const _Float16* __restrict__ w16, const _Float16* __restrict__ acts,
                          const float* __restrict__ rb, _Float16* __restrict__ aCur,
                          float* __restrict__ outAcc, int last) {
  int M0, N0; wave_tile_origin(M0, N0);
  int b = blockIdx.z;
  v8f c[2][4] = {};
  wmma_tile_2x4(w16, 256, acts + (size_t)b * TG * CH, 256, M0, N0, 256, c);
  int lane = threadIdx.x & 31;
  int nl = lane & 15, mh = (lane >> 4) << 3;
#pragma unroll
  for (int ni = 0; ni < 4; ++ni) {
    int t = N0 + ni * 16 + nl;
#pragma unroll
    for (int mi = 0; mi < 2; ++mi)
#pragma unroll
      for (int r = 0; r < 8; ++r) {
        int m = M0 + mi * 16 + mh + r;
        float d = c[mi][ni][r] + rb[m];
        if (last) {
          outAcc[((size_t)b * TG + t) * CH + m] += d;
        } else if (m < 256) {
          size_t i = ((size_t)b * TG + t) * CH + m;
          aCur[i] = (_Float16)((float)aCur[i] + d);
        } else {
          outAcc[((size_t)b * TG + t) * CH + (m - 256)] += d;
        }
      }
  }
}

// end 1x1 (M = 2*nh <= 8, K = 256) -> out2[b][2nh][t]
__global__ void k_end(const float* __restrict__ ew, const float* __restrict__ eb,
                      const float* __restrict__ outAcc, int nh2, float* __restrict__ out2) {
  int idx = blockIdx.x * blockDim.x + threadIdx.x;
  if (idx >= NB * nh2 * TG) return;
  int t = idx % TG;
  int o = (idx / TG) % nh2;
  int b = idx / (TG * nh2);
  float s = eb[o];
  const float* row = outAcc + ((size_t)b * TG + t) * CH;
  for (int c = 0; c < CH; ++c) s += ew[o * CH + c] * row[c];
  out2[idx] = s;
}

// affine coupling: a1 = exp(log_s)*a1 + bb ; emit log_s to d_out
__global__ void k_affine(const float* __restrict__ out2, int nh, int chBase,
                         float* __restrict__ aBuf, float* __restrict__ logsOut) {
  int idx = blockIdx.x * blockDim.x + threadIdx.x;
  if (idx >= NB * nh * TG) return;
  int t = idx % TG;
  int g = (idx / TG) % nh;
  int b = idx / (TG * nh);
  float bb = out2[((size_t)b * 2 * nh + g) * TG + t];
  float ls = out2[((size_t)b * 2 * nh + nh + g) * TG + t];
  size_t ai = (size_t)b * 8192 + (chBase + nh + g) * TG + t;
  aBuf[ai] = expf(ls) * aBuf[ai] + bb;
  logsOut[idx] = ls;            // (B, nh, Tg) row-major
}

// copy audio channels (early outputs / final) into d_out audio block
__global__ void k_copy_ch(const float* __restrict__ aBuf, int chBase, int nch,
                          int outChOff, float* __restrict__ audioOut) {
  int idx = blockIdx.x * blockDim.x + threadIdx.x;
  if (idx >= NB * nch * TG) return;
  int t = idx % TG;
  int g = (idx / TG) % nch;
  int b = idx / (TG * nch);
  audioOut[(size_t)b * 8192 + (outChOff + g) * TG + t] =
      aBuf[(size_t)b * 8192 + (chBase + g) * TG + t];
}

// log|det(W)| * B*Tg via single-thread partial-pivot elimination (n <= 8)
__global__ void k_logdet(const float* __restrict__ W, int n, float scale,
                         float* __restrict__ out) {
  if (threadIdx.x != 0 || blockIdx.x != 0) return;
  float M[64];
  for (int i = 0; i < n * n; ++i) M[i] = W[i];
  float la = 0.f;
  for (int c = 0; c < n; ++c) {
    int p = c; float best = fabsf(M[c * n + c]);
    for (int r = c + 1; r < n; ++r) { float v = fabsf(M[r * n + c]); if (v > best) { best = v; p = r; } }
    if (p != c) for (int j = 0; j < n; ++j) { float tmp = M[c*n+j]; M[c*n+j] = M[p*n+j]; M[p*n+j] = tmp; }
    float piv = M[c * n + c];
    la += logf(fabsf(piv));
    for (int r = c + 1; r < n; ++r) {
      float f = M[r * n + c] / piv;
      for (int j = c; j < n; ++j) M[r * n + j] -= f * M[c * n + j];
    }
  }
  *out = scale * la;
}

// ---------------------------------------------------------------------------
extern "C" void kernel_launch(void* const* d_in, const int* in_sizes, int n_in,
                              void* d_out, int out_size, void* d_ws, size_t ws_size,
                              hipStream_t stream) {
  (void)in_sizes; (void)n_in; (void)out_size; (void)ws_size;

  const float* spect = (const float*)d_in[0];
  const float* audio = (const float*)d_in[1];
  const float* upw   = (const float*)d_in[2];
  const float* upb   = (const float*)d_in[3];
  // flow leaves: W, start_w, start_b, cond_w, cond_b, end_w, end_b,
  //              in_w[0..7], in_b[0..7], rs_w[0..7], rs_b[0..7]  (39 per flow)
  auto FP = [&](int k, int off) { return (const float*)d_in[4 + k * 39 + off]; };

  // ---- workspace carve ----
  char* wsb = (char*)d_ws;
  size_t off = 0;
  auto carve = [&](size_t bytes) -> void* {
    void* p = wsb + off;
    off += (bytes + 255) & ~(size_t)255;
    return p;
  };
  _Float16* s16    = (_Float16*)carve((size_t)NB * TG * 640 * 2);
  float*    cond   = (float*)   carve((size_t)NB * TG * 4096 * 4);
  _Float16* cw16   = (_Float16*)carve((size_t)4096 * 640 * 2);
  _Float16* inw16  = (_Float16*)carve((size_t)512 * 768 * 2);
  _Float16* rsw16  = (_Float16*)carve((size_t)512 * 256 * 2);
  _Float16* imcol  = (_Float16*)carve((size_t)NB * TG * 768 * 2);
  _Float16* aCur   = (_Float16*)carve((size_t)NB * TG * CH * 2);
  _Float16* acts   = (_Float16*)carve((size_t)NB * TG * CH * 2);
  float*    inAct  = (float*)   carve((size_t)NB * TG * 512 * 4);
  float*    outAcc = (float*)   carve((size_t)NB * TG * CH * 4);
  float*    aA     = (float*)   carve((size_t)NB * 8192 * 4);
  float*    aB     = (float*)   carve((size_t)NB * 8192 * 4);
  float*    out2   = (float*)   carve((size_t)NB * 16 * TG * 4);

  float* outAudio  = (float*)d_out;                 // (B, 8, Tg)
  float* outLogs   = outAudio + NB * 8 * TG;        // 12 blocks of (B, nh_k, Tg)
  float* outLogdet = outAudio + NB * 8 * TG + 73728;

  auto nb = [](int n) { return (n + 255) / 256; };

  // conditioning input + audio regroup
  k_upsample  <<<nb(NB * 80 * 8192), 256, 0, stream>>>(spect, upw, upb, s16);
  k_audio_init<<<nb(NB * 8 * TG),   256, 0, stream>>>(audio, aA);

  float* aPing = aA;
  float* aPong = aB;
  int nh = 4, nrem = 8, chBase = 0, earlyOut = 0;
  size_t logsOff = 0;

  for (int k = 0; k < 12; ++k) {
    if ((k % 4) == 0 && k > 0) {                    // early output: peel 2 channels
      k_copy_ch<<<nb(NB * 2 * TG), 256, 0, stream>>>(aPing, chBase, 2, earlyOut, outAudio);
      earlyOut += 2; chBase += 2; nrem -= 2; nh -= 1;
    }

    const float* W = FP(k, 0);
    k_logdet<<<1, 1, 0, stream>>>(W, nrem, (float)(NB * TG), outLogdet + k);
    k_mmw   <<<nb(NB * nrem * TG), 256, 0, stream>>>(W, aPing, aPong, nrem, chBase);
    { float* t = aPing; aPing = aPong; aPong = t; }

    // cond = cond_w @ s  (WMMA, M=4096 K=640 N=1024)
    k_f2h<<<nb(4096 * 640), 256, 0, stream>>>(FP(k, 3), cw16, 4096 * 640);
    k_gemm_cond<<<dim3(32, 8, NB), 256, 0, stream>>>(cw16, s16, FP(k, 4), cond);

    // start 1x1 -> aCur (and zero skip accumulator)
    k_start<<<nb(NB * TG * CH), 256, 0, stream>>>(FP(k, 1), FP(k, 2), aPing, nh, chBase,
                                                  aCur, outAcc);

    for (int i = 0; i < 8; ++i) {
      k_conv_inw<<<nb(512 * 768), 256, 0, stream>>>(FP(k, 7 + i), inw16);
      k_im2col  <<<nb(NB * TG * 768), 256, 0, stream>>>(aCur, imcol, 1 << i);
      // in_act GEMM (WMMA, M=512 K=768 N=1024) with fused bias + cond add
      k_gemm_inact<<<dim3(4, 8, NB), 256, 0, stream>>>(inw16, imcol, FP(k, 15 + i),
                                                       cond, i * 512, inAct);
      k_act<<<nb(NB * TG * CH), 256, 0, stream>>>(inAct, acts);
      // res/skip GEMM (WMMA, M=512 or 256, K=256, N=1024) with fused residual
      int Mrs = (i < 7) ? 512 : 256;
      k_f2h<<<nb(Mrs * 256), 256, 0, stream>>>(FP(k, 23 + i), rsw16, Mrs * 256);
      k_gemm_rs<<<dim3(Mrs / 128, 8, NB), 256, 0, stream>>>(rsw16, acts, FP(k, 31 + i),
                                                            aCur, outAcc, (i == 7) ? 1 : 0);
    }

    // end 1x1 -> (log_s, b); affine coupling; emit log_s
    k_end   <<<nb(NB * 2 * nh * TG), 256, 0, stream>>>(FP(k, 5), FP(k, 6), outAcc, 2 * nh, out2);
    k_affine<<<nb(NB * nh * TG),     256, 0, stream>>>(out2, nh, chBase, aPing,
                                                       outLogs + logsOff);
    logsOff += (size_t)NB * nh * TG;
  }

  // final 4 remaining channels -> audio out channels 4..7
  k_copy_ch<<<nb(NB * nrem * TG), 256, 0, stream>>>(aPing, chBase, nrem, earlyOut, outAudio);
}
